// SoftNCutsLoss_36988258353193
// MI455X (gfx1250) — matrix operand order repaired and checked by reading, not compiled, not run on hardware
//
#include <hip/hip_runtime.h>
#include <hip/hip_bf16.h>

typedef __attribute__((ext_vector_type(16))) _Float16 v16h;
typedef __attribute__((ext_vector_type(8)))  float    v8f;

#define NB     2      // batches
#define KC     4      // classes
#define SS     8000   // H*W*D
#define TILE   32
#define NTILES (SS / TILE)   // 250
#define NPAIRS (NTILES * (NTILES + 1) / 2)   // 31375 triangular tile pairs
#define EPSV   1e-8f

// sqrt(log2(e)) : pre-scale x so exp(-(xi-xj)^2) == exp2(-(xi'-xj')^2)
#define XSCALE 1.2011224087864498f

// raw v_exp_f32 (flush-to-zero on underflow is fine here: exp(-large) ~ 0)
#if __has_builtin(__builtin_amdgcn_exp2f)
#define EXP2(xx) __builtin_amdgcn_exp2f(xx)
#else
#define EXP2(xx) exp2f(xx)
#endif

// ---------------------------------------------------------------------------
// Prep 1: pack A-operand fragments (16x32 f16) in contiguous-k order (same k order as the B fragments).
// Rows 0..3 = labels, row 4 = ones (column sums -> transposed-den term),
// rows 5..15 = zero.
//   lane l: M = l&15, khalf = l>>4
//   element e (0..15): K = khalf*16 + e
// ---------------------------------------------------------------------------
__global__ void sncuts_prep(const float* __restrict__ labels,
                            _Float16* __restrict__ afrag) {
    int idx = blockIdx.x * blockDim.x + threadIdx.x;
    const int total = NB * NTILES * 32;
    if (idx >= total) return;
    int lane  = idx & 31;
    int tile  = (idx >> 5) % NTILES;
    int n     = idx / (32 * NTILES);
    int m     = lane & 15;
    int khalf = lane >> 4;

    v16h vals;
#pragma unroll
    for (int e = 0; e < 16; ++e) {
        int i_local = khalf * 16 + e;
        int i = tile * TILE + i_local;
        float v = (m < KC) ? labels[((n * KC) + m) * SS + i]
                           : ((m == KC) ? 1.0f : 0.0f);
        vals[e] = (_Float16)v;
    }
    *(v16h*)&afrag[((size_t)(n * NTILES + tile) * 32 + lane) * 16] = vals;
}

// ---------------------------------------------------------------------------
// Prep 2: (a) pre-scaled x copy  xs[s] = x[s]*sqrt(log2 e)
//         (b) labels repacked as float4 per voxel: ajp[n*SS + s] = {A0..A3}[s]
//         (c) zero the 16 accumulator floats
// ---------------------------------------------------------------------------
__global__ void sncuts_prep2(const float* __restrict__ labels,
                             const float* __restrict__ x,
                             float* __restrict__ xs,
                             float4* __restrict__ ajp,
                             float* __restrict__ acc) {
    int idx = blockIdx.x * blockDim.x + threadIdx.x;
    if (idx < 16) acc[idx] = 0.0f;
    if (idx >= NB * SS) return;
    int n = idx / SS, s = idx % SS;
    xs[idx] = x[idx] * XSCALE;
    float4 a;
    a.x = labels[((n * KC) + 0) * SS + s];
    a.y = labels[((n * KC) + 1) * SS + s];
    a.z = labels[((n * KC) + 2) * SS + s];
    a.w = labels[((n * KC) + 3) * SS + s];
    ajp[idx] = a;
}

__device__ __forceinline__ int row_start(int I) {
    return I * NTILES - (I * (I - 1)) / 2;
}

// ---------------------------------------------------------------------------
// Prep 3: triangular pair table ptab[p] = (I<<16)|J so the hot loop replaces
// the sqrt+fixup decode with a single cached b32 load.
// ---------------------------------------------------------------------------
__global__ void sncuts_prep3(unsigned int* __restrict__ ptab) {
    int p = blockIdx.x * blockDim.x + threadIdx.x;
    if (p >= NPAIRS) return;
    int I = (int)((501.0f - __fsqrt_rn((float)(4 * NTILES * NTILES +
                   4 * NTILES + 1 - 8 * p))) * 0.5f);
    if (I < 0) I = 0;
    if (I > NTILES - 1) I = NTILES - 1;
    while (row_start(I) > p) --I;
    while (row_start(I + 1) <= p) ++I;
    int J = I + (p - row_start(I));
    ptab[p] = ((unsigned int)I << 16) | (unsigned int)J;
}

// ---------------------------------------------------------------------------
// Main: per wave, loop over triangular (I<=J) 32x32 tiles of the implicit
// affinity matrix W.  All operands come from direct global loads (WGP$/L2
// resident, ~1MB total footprint) -- no DS ops in the hot loop.
//   U = Afrag(I) x Wtile          (2x v_wmma_f32_16x16x32_f16)
//   num_k += sc * sum_j U[k,j]*A[k,j]        (sc = 1 diag, 2 off-diag)
//   den_k += sum_j U[k,j] + od * A[k,j]*U[4,j]   (U[4,*] = colsums, od = sc-1)
// acc layout: acc[n*8 + k] = num, acc[n*8 + 4 + k] = den.
// The pair loop is unrolled x2 so the scheduler can fill the WMMA->VALU
// hazard slots (4 co-exec NOPs each) with the other iteration's exp chain.
// ---------------------------------------------------------------------------
__global__ void sncuts_main(const float* __restrict__ xs,
                            const float4* __restrict__ ajp,
                            const _Float16* __restrict__ afrag,
                            const unsigned int* __restrict__ ptab,
                            float* __restrict__ acc) {
    const int n     = blockIdx.y;
    const int lane  = threadIdx.x & 31;
    const int wave  = (blockIdx.x * blockDim.x + threadIdx.x) >> 5;
    const int nwave = (gridDim.x * blockDim.x) >> 5;
    const int col   = lane & 15;
    const int ibase = (lane >> 4) * 16;  // B layout: lanes 0-15 -> i 0..15, 16-31 -> i 16..31

    const float*  xn  = xs  + n * SS;
    const float4* ajn = ajp + n * SS;

    float num[KC] = {0.f, 0.f, 0.f, 0.f};
    float den[KC] = {0.f, 0.f, 0.f, 0.f};

#pragma unroll 2
    for (int p = wave; p < NPAIRS; p += nwave) {
        unsigned int ij = ptab[p];
        int I = (int)(ij >> 16);
        int J = (int)(ij & 0xffffu);

        // 16 contiguous pre-scaled x values for this lane's i-half
        // (uniform per 16 lanes, 64B-aligned -> 4x b128 broadcast loads)
        float xv[16];
        {
            const float4* xp4 = (const float4*)(xn + I * TILE + ibase);
#pragma unroll
            for (int q = 0; q < 4; ++q) {
                float4 f = xp4[q];
                xv[4 * q + 0] = f.x;
                xv[4 * q + 1] = f.y;
                xv[4 * q + 2] = f.z;
                xv[4 * q + 3] = f.w;
            }
        }
        // per-lane j-column x values (one per j-half)
        float xj0 = xn[J * TILE + col];
        float xj1 = xn[J * TILE + 16 + col];

        // packed label columns of the J tile for the epilogue (2x b128)
        float4 a0q = ajn[J * TILE + col];
        float4 a1q = ajn[J * TILE + 16 + col];
        float a0[KC] = {a0q.x, a0q.y, a0q.z, a0q.w};
        float a1[KC] = {a1q.x, a1q.y, a1q.z, a1q.w};

        // prepacked A-operand fragment for tile I
        v16h A = *(const v16h*)&afrag[((size_t)(n * NTILES + I) * 32 + lane) * 16];

        const float sc = (I == J) ? 1.0f : 2.0f;   // off-diag tiles counted twice
        const float od = sc - 1.0f;                // transposed-den term enable

        // build both W-tile B-fragments together (SLP-friendly packed f32 math)
        v16h b0, b1;
#pragma unroll
        for (int t = 0; t < 16; ++t) {
            float d0 = xv[t] - xj0;
            float d1 = xv[t] - xj1;
            float m0 = -(d0 * d0);
            float m1 = -(d1 * d1);
            b0[t] = (_Float16)EXP2(m0);   // raw v_exp_f32
            b1[t] = (_Float16)EXP2(m1);
        }

        v8f u0 = {}, u1 = {};
        u0 = __builtin_amdgcn_wmma_f32_16x16x32_f16(
                false, A, false, b0, (short)0, u0, false, false);
        u1 = __builtin_amdgcn_wmma_f32_16x16x32_f16(
                false, A, false, b1, (short)0, u1, false, false);

        // lanes 0-15: u[r] = U[r, j], u[4] = colsum_j; lanes 16-31: rows 8+r == 0
        const float cs0 = od * u0[4];
        const float cs1 = od * u1[4];
#pragma unroll
        for (int r = 0; r < KC; ++r) {
            num[r] += sc * (u0[r] * a0[r] + u1[r] * a1[r]);
            den[r] += (u0[r] + u1[r]) + (a0[r] * cs0 + a1[r] * cs1);
        }
    }

    // cross-lane reduction (wave32)
#pragma unroll
    for (int r = 0; r < KC; ++r) {
        float nr = num[r], dr = den[r];
        for (int off = 16; off > 0; off >>= 1) {
            nr += __shfl_xor(nr, off, 32);
            dr += __shfl_xor(dr, off, 32);
        }
        if (lane == 0) {
            atomicAdd(&acc[n * 8 + r], nr);
            atomicAdd(&acc[n * 8 + 4 + r], dr);
        }
    }
}

__global__ void sncuts_final(const float* __restrict__ acc,
                             float* __restrict__ out) {
    int n = threadIdx.x;
    if (n < NB) {
        float s = 0.0f;
#pragma unroll
        for (int k = 0; k < KC; ++k)
            s += acc[n * 8 + k] / (acc[n * 8 + 4 + k] + EPSV);
        out[n] = (float)KC - s;
    }
}

extern "C" void kernel_launch(void* const* d_in, const int* in_sizes, int n_in,
                              void* d_out, int out_size, void* d_ws, size_t ws_size,
                              hipStream_t stream) {
    const float* labels = (const float*)d_in[0];   // (2,4,20,20,20) f32
    const float* x      = (const float*)d_in[1];   // (2,1,20,20,20) f32
    float* out          = (float*)d_out;           // (2,) f32

    // workspace layout:
    //   [0      , 512000)  A-fragments (2*250*32*16 f16)
    //   [512K   , +64000)  xs: pre-scaled x (2*8000 f32)
    //   [576K   , +256000) ajp: packed labels (2*8000 float4)
    //   [896K   , +64)     acc: 16 f32
    //   [900K   , +125500) ptab: triangular pair table (u32)
    _Float16*     afrag = (_Float16*)d_ws;
    float*        xsbuf = (float*)((char*)d_ws + 512 * 1024);
    float4*       ajp   = (float4*)((char*)d_ws + 576 * 1024);
    float*        acc   = (float*)((char*)d_ws + 896 * 1024);
    unsigned int* ptab  = (unsigned int*)((char*)d_ws + 900 * 1024);

    {   // pack label A-operand fragments (+ ones row for column sums)
        int total = NB * NTILES * 32;
        sncuts_prep<<<(total + 255) / 256, 256, 0, stream>>>(labels, afrag);
    }
    {   // pre-scaled x, packed epilogue labels, zeroed accumulators
        int total = NB * SS;
        sncuts_prep2<<<(total + 255) / 256, 256, 0, stream>>>(labels, x, xsbuf, ajp, acc);
    }
    sncuts_prep3<<<(NPAIRS + 255) / 256, 256, 0, stream>>>(ptab);
    {   // main triangular tile sweep: 512 blocks x 8 waves per batch
        dim3 grid(512, NB, 1);
        sncuts_main<<<grid, 256, 0, stream>>>(xsbuf, ajp, afrag, ptab, acc);
    }
    sncuts_final<<<1, 32, 0, stream>>>(acc, out);
}
